// ParsevalWaveletAttention_68204080661234
// MI455X (gfx1250) — compile-verified, matrix operand order checked by reading, not run on hardware
//
#include <hip/hip_runtime.h>
#include <hip/hip_bf16.h>
#include <math.h>

// ---------------- constants ----------------
constexpr int Bc  = 2;
constexpr int Tc  = 2048;
constexpr int Cc  = 1024;
constexpr int Hc  = 8;
constexpr int NBc = 32;          // T / 64 blocks
constexpr float GAMMA = 0.5f;

typedef __attribute__((ext_vector_type(16))) __bf16 v16bf;
typedef __attribute__((ext_vector_type(8)))  float  v8f;

// ---------------- CDNA5 async global->LDS staging ----------------
// Probe-verified signature: (int4 AS(1)*, int4 AS(3)*, imm offset, imm cpol)
#if defined(__AMDGCN__) && __has_builtin(__builtin_amdgcn_global_load_async_to_lds_b128) && \
    __has_builtin(__builtin_amdgcn_s_wait_asynccnt)
#define PWA_ASYNC 1
typedef int pwa_v4i __attribute__((vector_size(16)));
__device__ __forceinline__ void pwa_async_b128(const void* g, void* l) {
  __builtin_amdgcn_global_load_async_to_lds_b128(
      (__attribute__((address_space(1))) pwa_v4i*)g,
      (__attribute__((address_space(3))) pwa_v4i*)l, 0, 0);
}
__device__ __forceinline__ void pwa_async_wait() { __builtin_amdgcn_s_wait_asynccnt(0); }
#else
#define PWA_ASYNC 0
#endif

// ---------------- WMMA helpers (gfx1250 16x16x32 bf16) ----------------
// A/B operand element e -> K index (ISA 7.12.2, 16-bit A 16x32 layout)
__device__ __forceinline__ int wk_of(int e, int half) {
  return (e < 8) ? (half * 8 + e) : (16 + half * 8 + (e - 8));
}

__device__ __forceinline__ v8f wmma_bf16(v16bf a, v16bf b, v8f c) {
  return __builtin_amdgcn_wmma_f32_16x16x32_bf16(false, a, false, b, (short)0, c, false, false);
}

// operand stored row-major [row][k]; rows striped over lane%16
__device__ __forceinline__ v16bf frag_rowmajor(const __bf16* p, int stride, int row_base, int k_base) {
  const int l = threadIdx.x & 31;
  const int half = l >> 4, r = l & 15;
  const __bf16* rp = p + (size_t)(row_base + r) * stride + k_base;
  v16bf f;
#pragma unroll
  for (int e = 0; e < 16; ++e) f[e] = rp[wk_of(e, half)];
  return f;
}

// operand stored k-major [k][col] (B matrix for E@V)
__device__ __forceinline__ v16bf frag_kmajor(const __bf16* p, int stride, int k_base, int col_base) {
  const int l = threadIdx.x & 31;
  const int half = l >> 4, r = l & 15;
  v16bf f;
#pragma unroll
  for (int e = 0; e < 16; ++e) f[e] = p[(size_t)(k_base + wk_of(e, half)) * stride + col_base + r];
  return f;
}

// ---------------- simple elementwise / prep kernels ----------------
__global__ void pwa_cvt_bf16(const float* __restrict__ s, __bf16* __restrict__ d, int n) {
  int i = blockIdx.x * 256 + threadIdx.x;
  if (i < n) d[i] = (__bf16)s[i];
}

__global__ void pwa_prep_S(const float* __restrict__ A, float* __restrict__ S) {
  int idx = blockIdx.x * 256 + threadIdx.x;           // H*128*128
  int j = idx & 127, i = (idx >> 7) & 127, h = idx >> 14;
  S[idx] = A[(h * 128 + i) * 128 + j] - A[(h * 128 + j) * 128 + i];
}

__global__ __launch_bounds__(128) void pwa_prep_kn(const float* __restrict__ k_null, float* __restrict__ kn) {
  int h = blockIdx.x, d = threadIdx.x;
  __shared__ float red[128];
  float v = k_null[h * 128 + d];
  red[d] = v * v; __syncthreads();
  for (int s = 64; s > 0; s >>= 1) { if (d < s) red[d] += red[d + s]; __syncthreads(); }
  kn[h * 128 + d] = v / (sqrtf(red[0]) + 1e-6f);      // l2n eps
}

__global__ __launch_bounds__(64) void pwa_prep_wa(float* __restrict__ Wa) {
  int i = threadIdx.x;
  __shared__ float red[64];
  float t = -1.f + 2.f * (float)i / 63.f;
  red[i] = t * t; __syncthreads();
  for (int s = 32; s > 0; s >>= 1) { if (i < s) red[i] += red[i + s]; __syncthreads(); }
  float n1 = sqrtf(red[0]);
  Wa[i * 2 + 0] = 1.f / (8.f + 1e-6f);                // ||ones||=8
  Wa[i * 2 + 1] = t / (n1 + 1e-6f);
}

__global__ __launch_bounds__(256) void pwa_layernorm(const float* __restrict__ x,
    const float* __restrict__ w, const float* __restrict__ bb, __bf16* __restrict__ out) {
  __shared__ float red[256];
  const size_t bt = blockIdx.x;
  const int tid = threadIdx.x;
  const float* xp = x + bt * Cc;
  float s = 0.f;
  for (int c = tid; c < Cc; c += 256) s += xp[c];
  red[tid] = s; __syncthreads();
  for (int st = 128; st > 0; st >>= 1) { if (tid < st) red[tid] += red[tid + st]; __syncthreads(); }
  float mu = red[0] * (1.f / Cc);
  __syncthreads();
  float vs = 0.f;
  for (int c = tid; c < Cc; c += 256) { float d0 = xp[c] - mu; vs += d0 * d0; }
  red[tid] = vs; __syncthreads();
  for (int st = 128; st > 0; st >>= 1) { if (tid < st) red[tid] += red[tid + st]; __syncthreads(); }
  float rstd = rsqrtf(red[0] * (1.f / Cc) + 1e-5f);
  for (int c = tid; c < Cc; c += 256)
    out[bt * Cc + c] = (__bf16)((xp[c] - mu) * rstd * w[c] + bb[c]);
}

// ---------------- GEMM: C[M,N] = A[M,K] * B[N,K]^T (bf16 in, f32 out) ----------------
__global__ __launch_bounds__(256) void pwa_gemm_nt(const __bf16* __restrict__ A,
    const __bf16* __restrict__ Bm, float* __restrict__ C, int M, int N, int Kd) {
  __shared__ __bf16 sA[128 * 40];
  __shared__ __bf16 sB[128 * 40];
  const int m0 = blockIdx.x * 128, n0 = blockIdx.y * 128;
  const int tid = threadIdx.x;
  const int w = tid >> 5, l = tid & 31;
  const int rg = w >> 1, cg = w & 1;
  const int half = l >> 4, r16 = l & 15;
  v8f acc[2][4] = {};
  for (int k0 = 0; k0 < Kd; k0 += 32) {
#if PWA_ASYNC
#pragma unroll
    for (int it = tid; it < 512; it += 256) {
      int rr = it >> 2, sg = it & 3;
      pwa_async_b128(A + (size_t)(m0 + rr) * Kd + k0 + sg * 8, &sA[rr * 40 + sg * 8]);
      pwa_async_b128(Bm + (size_t)(n0 + rr) * Kd + k0 + sg * 8, &sB[rr * 40 + sg * 8]);
    }
    pwa_async_wait();
#else
#pragma unroll
    for (int it = tid; it < 512; it += 256) {
      int rr = it >> 2, sg = it & 3;
      *(uint4*)(&sA[rr * 40 + sg * 8]) = *(const uint4*)(A + (size_t)(m0 + rr) * Kd + k0 + sg * 8);
      *(uint4*)(&sB[rr * 40 + sg * 8]) = *(const uint4*)(Bm + (size_t)(n0 + rr) * Kd + k0 + sg * 8);
    }
#endif
    __syncthreads();
    v16bf af[2], bfr[4];
#pragma unroll
    for (int mi = 0; mi < 2; ++mi) af[mi] = frag_rowmajor(sA, 40, rg * 32 + mi * 16, 0);
#pragma unroll
    for (int ni = 0; ni < 4; ++ni) bfr[ni] = frag_rowmajor(sB, 40, cg * 64 + ni * 16, 0);
#pragma unroll
    for (int mi = 0; mi < 2; ++mi)
#pragma unroll
      for (int ni = 0; ni < 4; ++ni)
        acc[mi][ni] = wmma_bf16(af[mi], bfr[ni], acc[mi][ni]);
    __syncthreads();
  }
#pragma unroll
  for (int mi = 0; mi < 2; ++mi)
#pragma unroll
    for (int ni = 0; ni < 4; ++ni)
#pragma unroll
      for (int v = 0; v < 8; ++v) {
        int row = m0 + rg * 32 + mi * 16 + half * 8 + v;
        int col = n0 + cg * 64 + ni * 16 + r16;
        C[(size_t)row * N + col] = acc[mi][ni][v];
      }
}

// ---------------- RoPE + l2n, (B,T,H,DH) -> Qcat/Kcat[...,0:128], null score ----------------
__global__ __launch_bounds__(256) void pwa_rope_l2n(const float* __restrict__ raw,
    const float* __restrict__ kn, __bf16* __restrict__ cat, float* __restrict__ nullsc, int isQ) {
  int gidx = blockIdx.x * 8 + (threadIdx.x >> 5);     // (b*T+t)*H + h
  int l = threadIdx.x & 31;
  int h = gidx % Hc; int bt = gidx / Hc;
  int t = bt % Tc;   int b = bt / Tc;
  const float* src = raw + (size_t)bt * Cc + h * 128;
  float v0 = src[l], v1 = src[l + 32], v2 = src[l + 64], v3 = src[l + 96];
  float a0 = (float)t * powf(10000.f, -((float)l) / 64.f);
  float a1 = (float)t * powf(10000.f, -((float)(l + 32)) / 64.f);
  float c0 = cosf(a0), s0 = sinf(a0), c1 = cosf(a1), s1 = sinf(a1);
  float o0 = v0 * c0 - v2 * s0;
  float o2 = v0 * s0 + v2 * c0;
  float o1 = v1 * c1 - v3 * s1;
  float o3 = v1 * s1 + v3 * c1;
  float sq = o0 * o0 + o1 * o1 + o2 * o2 + o3 * o3;
#pragma unroll
  for (int off = 16; off > 0; off >>= 1) sq += __shfl_xor(sq, off);
  float inv = 1.f / (sqrtf(sq) + 1e-6f);
  o0 *= inv; o1 *= inv; o2 *= inv; o3 *= inv;
  __bf16* dst = cat + (((size_t)(b * Hc + h)) * Tc + t) * 256;
  dst[l] = (__bf16)o0; dst[l + 32] = (__bf16)o1;
  dst[l + 64] = (__bf16)o2; dst[l + 96] = (__bf16)o3;
  if (isQ) {
    const float* knh = kn + h * 128;
    float p = o0 * knh[l] + o1 * knh[l + 32] + o2 * knh[l + 64] + o3 * knh[l + 96];
#pragma unroll
    for (int off = 16; off > 0; off >>= 1) p += __shfl_xor(p, off);
    if (l == 0) nullsc[((size_t)(b * Hc + h)) * Tc + t] = p;
  }
}

__global__ void pwa_vpost(const float* __restrict__ raw, __bf16* __restrict__ Vbf) {
  int idx = blockIdx.x * 256 + threadIdx.x;            // B*T*C
  int c = idx & (Cc - 1); int bt = idx >> 10;
  int h = c >> 7, d = c & 127;
  int b = bt / Tc, t = bt % Tc;
  Vbf[(((size_t)(b * Hc + h)) * Tc + t) * 128 + d] = (__bf16)raw[idx];
}

// xv = fnorm(x per head); Qcat[...,128:256] = gamma*(xv@S), Kcat[...,128:256] = xv
__global__ __launch_bounds__(128) void pwa_xs(const float* __restrict__ x, const float* __restrict__ S,
    __bf16* __restrict__ Qcat, __bf16* __restrict__ Kcat) {
  int t = blockIdx.x % Tc; int h = (blockIdx.x / Tc) % Hc; int b = blockIdx.x / (Tc * Hc);
  int d = threadIdx.x;
  __shared__ float sxv[128];
  __shared__ float red[128];
  float xv = x[((size_t)b * Tc + t) * Cc + h * 128 + d];
  red[d] = xv * xv; __syncthreads();
  for (int s = 64; s > 0; s >>= 1) { if (d < s) red[d] += red[d + s]; __syncthreads(); }
  float nrm = fmaxf(sqrtf(red[0]), 1e-12f);
  float xvn = xv / nrm;
  sxv[d] = xvn; __syncthreads();
  const float* Sh = S + (size_t)h * 128 * 128;
  float acc = 0.f;
  for (int e = 0; e < 128; ++e) acc += sxv[e] * Sh[e * 128 + d];
  size_t base = (((size_t)(b * Hc + h)) * Tc + t) * 256;
  Qcat[base + 128 + d] = (__bf16)(GAMMA * acc);
  Kcat[base + 128 + d] = (__bf16)xvn;
}

// wavelet transforms qf/kf/vf[b,h,n*2+kk,d]
__global__ void pwa_wavelet(const __bf16* __restrict__ Qcat, const __bf16* __restrict__ Kcat,
    const __bf16* __restrict__ Vbf, const float* __restrict__ Wa,
    float* __restrict__ qf, float* __restrict__ kf, float* __restrict__ vf) {
  int idx = blockIdx.x * 256 + threadIdx.x;            // B*H*NB*2*128
  int d = idx & 127; int kk = (idx >> 7) & 1; int nn = (idx >> 8) % NBc; int bh = idx / (128 * 2 * NBc);
  float aq = 0.f, ak = 0.f, av = 0.f;
  size_t tb = ((size_t)bh * Tc + nn * 64);
  for (int i = 0; i < 64; ++i) {
    float w = Wa[i * 2 + kk];
    aq += w * (float)Qcat[(tb + i) * 256 + d];
    ak += w * (float)Kcat[(tb + i) * 256 + d];
    av += w * (float)Vbf[(tb + i) * 128 + d];
  }
  size_t o = (((size_t)bh * NBc + nn) * 2 + kk) * 128 + d;
  qf[o] = aq; kf[o] = ak; vf[o] = av;
}

// vn = fnorm(vf); tS = vn @ S
__global__ __launch_bounds__(128) void pwa_vn_ts(const float* __restrict__ vf, const float* __restrict__ S,
    float* __restrict__ vn, float* __restrict__ tS) {
  int I = blockIdx.x & 63; int bh = blockIdx.x >> 6;
  int d = threadIdx.x;
  __shared__ float sv[128];
  __shared__ float red[128];
  size_t o = ((size_t)bh * 64 + I) * 128;
  float v = vf[o + d];
  red[d] = v * v; __syncthreads();
  for (int s = 64; s > 0; s >>= 1) { if (d < s) red[d] += red[d + s]; __syncthreads(); }
  float nrm = fmaxf(sqrtf(red[0]), 1e-12f);
  float vnv = v / nrm;
  vn[o + d] = vnv; sv[d] = vnv; __syncthreads();
  int h = bh % Hc;
  const float* Sh = S + (size_t)h * 128 * 128;
  float acc = 0.f;
  for (int e = 0; e < 128; ++e) acc += sv[e] * Sh[e * 128 + d];
  tS[o + d] = acc;
}

// fc[bh,I,J] = qf_I . kf_J + gamma * tS_I . vn_J
__global__ void pwa_fc(const float* __restrict__ qf, const float* __restrict__ kf,
    const float* __restrict__ tS, const float* __restrict__ vn, float* __restrict__ fc) {
  int idx = blockIdx.x * 256 + threadIdx.x;            // B*H*64*64
  int J = idx & 63; int I = (idx >> 6) & 63; int bh = idx >> 12;
  const float* qI = qf + ((size_t)bh * 64 + I) * 128;
  const float* kJ = kf + ((size_t)bh * 64 + J) * 128;
  const float* tI = tS + ((size_t)bh * 64 + I) * 128;
  const float* vJ = vn + ((size_t)bh * 64 + J) * 128;
  float s = 0.f, s2 = 0.f;
  for (int d = 0; d < 128; ++d) { s += qI[d] * kJ[d]; s2 += tI[d] * vJ[d]; }
  fc[idx] = s + GAMMA * s2;
}

// ---------------- row statistics for vss (sum/sumsq over finite scores) ----------------
__global__ __launch_bounds__(256) void pwa_stats(const __bf16* __restrict__ Qcat,
    const __bf16* __restrict__ Kcat, const float* __restrict__ fc, const float* __restrict__ Wa,
    const float* __restrict__ nullsc, float* __restrict__ meanArr, float* __restrict__ rstdArr) {
  const int n = blockIdx.x, h = blockIdx.y, b = blockIdx.z;
  const int bh = b * Hc + h;
  const int tid = threadIdx.x;
  const int r = tid >> 2, q4 = tid & 3;
  const int i = n * 64 + r;
  const size_t gi = (size_t)bh * Tc + i;
  __shared__ float sWa0[64], sWa1[64];
  __shared__ __bf16 sQ[64 * 256];
  if (tid < 64) { sWa0[tid] = Wa[tid * 2]; sWa1[tid] = Wa[tid * 2 + 1]; }
  {
    const __bf16* src = Qcat + ((size_t)bh * Tc + n * 64) * 256;
#if PWA_ASYNC
    for (int it = tid; it < 2048; it += 256) pwa_async_b128(src + it * 8, &sQ[it * 8]);
    pwa_async_wait();
#else
    for (int it = tid; it < 2048; it += 256)
      *(uint4*)(&sQ[it * 8]) = ((const uint4*)src)[it];
#endif
  }
  __syncthreads();
  const __bf16* qrow = &sQ[r * 256];
  float sum = 0.f, ssq = 0.f;
  // near band j in [max(0,i-64), i]
  int lo = (i > 64) ? (i - 64) : 0;
  for (int j = lo + q4; j <= i; j += 4) {
    const __bf16* kp = Kcat + ((size_t)bh * Tc + j) * 256;
    float s = 0.f;
#pragma unroll 8
    for (int d = 0; d < 256; ++d) s += (float)qrow[d] * (float)kp[d];
    sum += s; ssq += s * s;
  }
  // far: full blocks 0..n-2
  const float wi0 = sWa0[r], wi1 = sWa1[r];
  for (int m = 0; m < n - 1; ++m) {
    const float* g = fc + ((size_t)bh * 64 + n * 2) * 64 + m * 2;
    float a0 = wi0 * g[0] + wi1 * g[64];
    float a1 = wi0 * g[1] + wi1 * g[65];
    for (int jl = q4; jl < 64; jl += 4) {
      float s = a0 * sWa0[jl] + a1 * sWa1[jl];
      sum += s; ssq += s * s;
    }
  }
  // far partial in block n-1: jl < r
  if (n >= 1) {
    const float* g = fc + ((size_t)bh * 64 + n * 2) * 64 + (n - 1) * 2;
    float a0 = wi0 * g[0] + wi1 * g[64];
    float a1 = wi0 * g[1] + wi1 * g[65];
    for (int jl = q4; jl < r; jl += 4) {
      float s = a0 * sWa0[jl] + a1 * sWa1[jl];
      sum += s; ssq += s * s;
    }
  }
  if (q4 == 0) { float s = nullsc[gi]; sum += s; ssq += s * s; }
  sum += __shfl_xor(sum, 1); sum += __shfl_xor(sum, 2);
  ssq += __shfl_xor(ssq, 1); ssq += __shfl_xor(ssq, 2);
  if (q4 == 0) {
    float cnt = (float)(i + 2);
    float mean = sum / cnt;
    float var = ssq / cnt - mean * mean;
    float std = sqrtf(fmaxf(var, 1e-6f));
    meanArr[gi] = mean; rstdArr[gi] = 1.f / std;
  }
}

// ---------------- flash pass: exp-weight scores, accumulate Y = E @ V ----------------
__global__ __launch_bounds__(256) void pwa_flash(const __bf16* __restrict__ Qcat,
    const __bf16* __restrict__ Kcat, const __bf16* __restrict__ Vbf,
    const float* __restrict__ fc, const float* __restrict__ Wa,
    const float* __restrict__ meanArr, const float* __restrict__ rstdArr,
    const float* __restrict__ nullsc, __bf16* __restrict__ ybf) {
  const int n = blockIdx.x, h = blockIdx.y, b = blockIdx.z;
  const int bh = b * Hc + h;
  const int tid = threadIdx.x;
  const int l = tid & 31, w = tid >> 5;
  const int half = l >> 4, r16 = l & 15;
  const int rg = w >> 1, cg = w & 1;

  __shared__ __bf16 sKV[64 * 264];   // Kcat tile (stride 264) reused as V tile (stride 136)
  __shared__ __bf16 sE[64 * 66];     // exp-weight tile, padded (33 dwords/row)
  __shared__ float sDen[64];
  __shared__ float sMean[64], sRstd[64];
  __shared__ float sWa0[64], sWa1[64];
  __shared__ float sG[4];

  const __bf16* Qrow = Qcat + ((size_t)bh * Tc + n * 64) * 256;
  if (tid < 64) {
    size_t gi = (size_t)bh * Tc + n * 64 + tid;
    sMean[tid] = meanArr[gi]; sRstd[tid] = rstdArr[gi];
    sDen[tid] = 0.f;
    sWa0[tid] = Wa[tid * 2]; sWa1[tid] = Wa[tid * 2 + 1];
  }
  __syncthreads();

  v8f yacc[4] = {};
  float denAcc[8] = {0.f, 0.f, 0.f, 0.f, 0.f, 0.f, 0.f, 0.f};

  for (int m = 0; m <= n; ++m) {
    __syncthreads();                         // protect sE/sKV from previous phase-2 reads
    if (tid < 4)
      sG[tid] = fc[((size_t)bh * 64 + n * 2 + (tid >> 1)) * 64 + m * 2 + (tid & 1)];
    if (m < n)                               // warm L2 for next V tile (global_prefetch_b8)
      __builtin_prefetch((const void*)(Vbf + ((size_t)bh * Tc + (m + 1) * 64) * 128), 0, 1);
    const bool nearTile = (m >= n - 1);
    if (nearTile) {
      // stage Kcat tile (64 x 256 bf16), padded stride 264
      const __bf16* src = Kcat + ((size_t)bh * Tc + m * 64) * 256;
#if PWA_ASYNC
      for (int it = tid; it < 2048; it += 256) {
        int rr = it >> 5, sg = it & 31;
        pwa_async_b128(src + rr * 256 + sg * 8, &sKV[rr * 264 + sg * 8]);
      }
      pwa_async_wait();
#else
      for (int it = tid; it < 2048; it += 256) {
        int rr = it >> 5, sg = it & 31;
        *(uint4*)(&sKV[rr * 264 + sg * 8]) = *(const uint4*)(src + rr * 256 + sg * 8);
      }
#endif
      __syncthreads();
      // S = Qcat_blk @ Kcat_blk^T (64x64 over 256), wave: rows rg*16.., cols cg*32..
      v8f sf[2] = {};
#pragma unroll
      for (int ks = 0; ks < 8; ++ks) {
        v16bf af;
        {
          const __bf16* ap = Qrow + (size_t)(rg * 16 + r16) * 256 + ks * 32;
#pragma unroll
          for (int e = 0; e < 16; ++e) af[e] = ap[wk_of(e, half)];
        }
#pragma unroll
        for (int ns = 0; ns < 2; ++ns) {
          v16bf bg = frag_rowmajor(sKV, 264, cg * 32 + ns * 16, ks * 32);
          sf[ns] = wmma_bf16(af, bg, sf[ns]);
        }
      }
      // mask / far-override / standardize / exponentiate
#pragma unroll
      for (int ns = 0; ns < 2; ++ns)
#pragma unroll
        for (int v = 0; v < 8; ++v) {
          int il = rg * 16 + half * 8 + v;
          int jl = cg * 32 + ns * 16 + r16;
          float e;
          if (m == n && jl > il) {
            e = 0.f;                          // causal mask inside diagonal block
          } else {
            float s = sf[ns][v];
            if (m == n - 1 && il > jl) {      // |i-j| > 64 -> rank-2 far value
              float a0 = sWa0[il] * sG[0] + sWa1[il] * sG[2];
              float a1 = sWa0[il] * sG[1] + sWa1[il] * sG[3];
              s = a0 * sWa0[jl] + a1 * sWa1[jl];
            }
            float z = (s - sMean[il]) * sRstd[il];
            e = __expf(fminf(z, 80.f));
          }
          sE[il * 66 + jl] = (__bf16)e;
          denAcc[v] += e;
        }
    } else {
      __syncthreads();                       // make sG visible
      // far-only tile: rank-2 evaluation, 4 threads per row
      int il = tid >> 2, q4 = tid & 3;
      float a0 = sWa0[il] * sG[0] + sWa1[il] * sG[2];
      float a1 = sWa0[il] * sG[1] + sWa1[il] * sG[3];
      float mean = sMean[il], rs = sRstd[il];
      float psum = 0.f;
      for (int jl = q4; jl < 64; jl += 4) {
        float s = a0 * sWa0[jl] + a1 * sWa1[jl];
        float e = __expf(fminf((s - mean) * rs, 80.f));
        sE[il * 66 + jl] = (__bf16)e;
        psum += e;
      }
      atomicAdd(&sDen[il], psum);
    }
    // phase 2: Y += E @ V
    __syncthreads();
    {
      const __bf16* vsrc = Vbf + ((size_t)bh * Tc + m * 64) * 128;
#if PWA_ASYNC
      for (int it = tid; it < 1024; it += 256) {
        int rr = it >> 4, sg = it & 15;
        pwa_async_b128(vsrc + rr * 128 + sg * 8, &sKV[rr * 136 + sg * 8]);
      }
      pwa_async_wait();
#else
      for (int it = tid; it < 1024; it += 256) {
        int rr = it >> 4, sg = it & 15;
        *(uint4*)(&sKV[rr * 136 + sg * 8]) = *(const uint4*)(vsrc + rr * 128 + sg * 8);
      }
#endif
    }
    __syncthreads();
#pragma unroll
    for (int ks = 0; ks < 2; ++ks) {
      v16bf af = frag_rowmajor(sE, 66, rg * 16, ks * 32);
#pragma unroll
      for (int ns = 0; ns < 4; ++ns) {
        v16bf bg = frag_kmajor(sKV, 136, ks * 32, cg * 64 + ns * 16);
        yacc[ns] = wmma_bf16(af, bg, yacc[ns]);
      }
    }
  }

  // fold per-lane denominators + null column, then normalize & store
#pragma unroll
  for (int v = 0; v < 8; ++v) atomicAdd(&sDen[rg * 16 + half * 8 + v], denAcc[v]);
  __syncthreads();
  if (tid < 64) {
    size_t gi = (size_t)bh * Tc + n * 64 + tid;
    float z = (nullsc[gi] - sMean[tid]) * sRstd[tid];
    sDen[tid] += __expf(fminf(z, 80.f));
  }
  __syncthreads();
#pragma unroll
  for (int ns = 0; ns < 4; ++ns)
#pragma unroll
    for (int v = 0; v < 8; ++v) {
      int il = rg * 16 + half * 8 + v;
      int col = cg * 64 + ns * 16 + r16;
      float val = yacc[ns][v] / sDen[il];
      int t = n * 64 + il;
      ybf[((size_t)b * Tc + t) * Cc + h * 128 + col] = (__bf16)val;
    }
}

// ---------------- host driver ----------------
static inline char* ws_take(char*& p, size_t bytes) {
  char* r = p;
  p += (bytes + 255) & ~(size_t)255;
  return r;
}

extern "C" void kernel_launch(void* const* d_in, const int* in_sizes, int n_in,
                              void* d_out, int out_size, void* d_ws, size_t ws_size,
                              hipStream_t stream) {
  (void)in_sizes; (void)n_in; (void)out_size; (void)ws_size;
  const float* x      = (const float*)d_in[0];
  const float* WQ     = (const float*)d_in[1];
  const float* WK     = (const float*)d_in[2];
  const float* WV     = (const float*)d_in[3];
  const float* WO     = (const float*)d_in[4];
  const float* ln_w   = (const float*)d_in[5];
  const float* ln_b   = (const float*)d_in[6];
  const float* A      = (const float*)d_in[7];
  const float* k_null = (const float*)d_in[8];
  float* out = (float*)d_out;

  const size_t NT = (size_t)Bc * Tc;          // 4096 rows
  const size_t NE = NT * Cc;                  // 4 M elements
  const size_t WE = (size_t)Cc * Cc;          // 1 M elements

  char* p = (char*)d_ws;
  __bf16* wq_bf  = (__bf16*)ws_take(p, WE * 2);
  __bf16* wk_bf  = (__bf16*)ws_take(p, WE * 2);
  __bf16* wv_bf  = (__bf16*)ws_take(p, WE * 2);
  __bf16* wo_bf  = (__bf16*)ws_take(p, WE * 2);
  __bf16* xbf    = (__bf16*)ws_take(p, NE * 2);
  __bf16* xlnbf  = (__bf16*)ws_take(p, NE * 2);
  float*  raw    = (float*) ws_take(p, NE * 4);
  __bf16* Qcat   = (__bf16*)ws_take(p, (size_t)Bc * Hc * Tc * 256 * 2);
  __bf16* Kcat   = (__bf16*)ws_take(p, (size_t)Bc * Hc * Tc * 256 * 2);
  __bf16* Vbf    = (__bf16*)ws_take(p, (size_t)Bc * Hc * Tc * 128 * 2);
  float*  Sbuf   = (float*) ws_take(p, (size_t)Hc * 128 * 128 * 4);
  float*  kn     = (float*) ws_take(p, (size_t)Hc * 128 * 4);
  float*  Wa     = (float*) ws_take(p, 64 * 2 * 4);
  float*  nullsc = (float*) ws_take(p, (size_t)Bc * Hc * Tc * 4);
  float*  qf     = (float*) ws_take(p, (size_t)Bc * Hc * 64 * 128 * 4);
  float*  kf     = (float*) ws_take(p, (size_t)Bc * Hc * 64 * 128 * 4);
  float*  vf     = (float*) ws_take(p, (size_t)Bc * Hc * 64 * 128 * 4);
  float*  vn     = (float*) ws_take(p, (size_t)Bc * Hc * 64 * 128 * 4);
  float*  tS     = (float*) ws_take(p, (size_t)Bc * Hc * 64 * 128 * 4);
  float*  fc     = (float*) ws_take(p, (size_t)Bc * Hc * 64 * 64 * 4);
  float*  meanA  = (float*) ws_take(p, (size_t)Bc * Hc * Tc * 4);
  float*  rstdA  = (float*) ws_take(p, (size_t)Bc * Hc * Tc * 4);
  __bf16* ybf    = (__bf16*)ws_take(p, NE * 2);

  // 1. conversions + prep
  pwa_cvt_bf16<<<(int)((WE + 255) / 256), 256, 0, stream>>>(WQ, wq_bf, (int)WE);
  pwa_cvt_bf16<<<(int)((WE + 255) / 256), 256, 0, stream>>>(WK, wk_bf, (int)WE);
  pwa_cvt_bf16<<<(int)((WE + 255) / 256), 256, 0, stream>>>(WV, wv_bf, (int)WE);
  pwa_cvt_bf16<<<(int)((WE + 255) / 256), 256, 0, stream>>>(WO, wo_bf, (int)WE);
  pwa_cvt_bf16<<<(int)((NE + 255) / 256), 256, 0, stream>>>(x, xbf, (int)NE);
  pwa_prep_S<<<(Hc * 128 * 128) / 256, 256, 0, stream>>>(A, Sbuf);
  pwa_prep_kn<<<Hc, 128, 0, stream>>>(k_null, kn);
  pwa_prep_wa<<<1, 64, 0, stream>>>(Wa);
  pwa_layernorm<<<(int)NT, 256, 0, stream>>>(x, ln_w, ln_b, xlnbf);

  // 2. projections (WMMA GEMMs) + per-head post
  dim3 gg((int)(NT / 128), Cc / 128);
  pwa_gemm_nt<<<gg, 256, 0, stream>>>(xbf, wq_bf, raw, (int)NT, Cc, Cc);
  pwa_rope_l2n<<<(int)(NT * Hc / 8), 256, 0, stream>>>(raw, kn, Qcat, nullsc, 1);
  pwa_gemm_nt<<<gg, 256, 0, stream>>>(xbf, wk_bf, raw, (int)NT, Cc, Cc);
  pwa_rope_l2n<<<(int)(NT * Hc / 8), 256, 0, stream>>>(raw, kn, Kcat, nullsc, 0);
  pwa_gemm_nt<<<gg, 256, 0, stream>>>(xlnbf, wv_bf, raw, (int)NT, Cc, Cc);
  pwa_vpost<<<(int)(NE / 256), 256, 0, stream>>>(raw, Vbf);
  pwa_xs<<<Bc * Hc * Tc, 128, 0, stream>>>(x, Sbuf, Qcat, Kcat);

  // 3. wavelet / far-field grid
  pwa_wavelet<<<(Bc * Hc * NBc * 2 * 128) / 256, 256, 0, stream>>>(Qcat, Kcat, Vbf, Wa, qf, kf, vf);
  pwa_vn_ts<<<Bc * Hc * 64, 128, 0, stream>>>(vf, Sbuf, vn, tS);
  pwa_fc<<<(Bc * Hc * 64 * 64) / 256, 256, 0, stream>>>(qf, kf, tS, vn, fc);

  // 4. vss row statistics, then fused flash attention
  dim3 ga(NBc, Hc, Bc);
  pwa_stats<<<ga, 256, 0, stream>>>(Qcat, Kcat, fc, Wa, nullsc, meanA, rstdA);
  pwa_flash<<<ga, 256, 0, stream>>>(Qcat, Kcat, Vbf, fc, Wa, meanA, rstdA, nullsc, ybf);

  // 5. output projection
  pwa_gemm_nt<<<gg, 256, 0, stream>>>(ybf, wo_bf, out, (int)NT, Cc, Cc);
}
